// NeuralOT_33002528703071
// MI455X (gfx1250) — matrix operand order repaired and checked by reading, not compiled, not run on hardware
//
#include <hip/hip_runtime.h>
#include <hip/hip_bf16.h>
#include <stdint.h>

typedef __attribute__((ext_vector_type(16))) __bf16 v16bf;
typedef __attribute__((ext_vector_type(8)))  __bf16 v8bf;
typedef __attribute__((ext_vector_type(8)))  float  v8f;
typedef int v4i __attribute__((vector_size(16)));   // matches builtin param type

#define OT_EPS      0.01f
#define OT_INV_EPS  100.0f
#define DD          3072
#define NROW        4096
#define BM          128
#define BN          128
#define BK          32
#define LDSW        40                // bf16 per LDS row (80B stride: 16B-aligned, conflict-free)
#define TILE_ELEMS  (BM * LDSW)       // bf16 elems per array per buffer
#define SHMEM_BYTES (2 * 4 * TILE_ELEMS * sizeof(__bf16) + 256 * sizeof(float))

#define AS1 __attribute__((address_space(1)))
#define AS3 __attribute__((address_space(3)))

// ---- CDNA5 async global->LDS path (ASYNCcnt), with sync fallback ----
#if __has_builtin(__builtin_amdgcn_global_load_async_to_lds_b128) && \
    __has_builtin(__builtin_amdgcn_s_wait_asynccnt)
#define OT_ASYNC 1
static __device__ __forceinline__ void ot_cp16(void* lds, const void* g) {
  __builtin_amdgcn_global_load_async_to_lds_b128(
      (AS1 v4i*)const_cast<void*>(g), (AS3 v4i*)lds, 0, 0);
}
static __device__ __forceinline__ void ot_cp_wait() {
  __builtin_amdgcn_s_wait_asynccnt(0);
}
#else
#define OT_ASYNC 0
static __device__ __forceinline__ void ot_cp16(void* lds, const void* g) {
  *(uint4*)lds = *(const uint4*)g;
}
static __device__ __forceinline__ void ot_cp_wait() {}
#endif

static __device__ __forceinline__ v16bf combine8(v8bf lo, v8bf hi) {
  return __builtin_shufflevector(lo, hi, 0,1,2,3,4,5,6,7,8,9,10,11,12,13,14,15);
}

// One block per row: emit bf16 hi/lo split of the row, and reduce
// pot = row . w + b[0], sq = sum(row^2).
__global__ __launch_bounds__(256)
void ot_prep_kernel(const float* __restrict__ src,
                    const float* __restrict__ w,
                    const float* __restrict__ b,
                    __bf16* __restrict__ hi,
                    __bf16* __restrict__ lo,
                    float* __restrict__ pot,
                    float* __restrict__ sq)
{
  __shared__ float s_dot[256];
  __shared__ float s_sq[256];
  const int row = blockIdx.x;
  const int t = threadIdx.x;
  const float* xr = src + (size_t)row * DD;
  __bf16* hr = hi + (size_t)row * DD;
  __bf16* lr = lo + (size_t)row * DD;
  float pd = 0.0f, ps = 0.0f;
  for (int j = t; j < DD; j += 256) {
    float xv = xr[j];
    __bf16 h = (__bf16)xv;
    float hf = (float)h;
    __bf16 l = (__bf16)(xv - hf);
    hr[j] = h;
    lr[j] = l;
    pd += xv * w[j];
    ps += xv * xv;
  }
  s_dot[t] = pd; s_sq[t] = ps;
  __syncthreads();
  for (int o = 128; o > 0; o >>= 1) {
    if (t < o) { s_dot[t] += s_dot[t + o]; s_sq[t] += s_sq[t + o]; }
    __syncthreads();
  }
  if (t == 0) { pot[row] = s_dot[0] + b[0]; sq[row] = s_sq[0]; }
}

// 128x128 tile GEMM (G = X @ Y^T), bf16x3 split accumulation in f32 WMMA,
// double-buffered async global->LDS pipeline, fused entropic-OT epilogue,
// per-block partial sums (deterministic, no FP atomics).
__global__ __launch_bounds__(256)
void ot_gemm_kernel(const __bf16* __restrict__ xhi, const __bf16* __restrict__ xlo,
                    const __bf16* __restrict__ yhi, const __bf16* __restrict__ ylo,
                    const float* __restrict__ u,  const float* __restrict__ vv,
                    const float* __restrict__ x2, const float* __restrict__ y2,
                    float* __restrict__ partials)
{
  extern __shared__ __attribute__((aligned(16))) char smem[];
  __bf16* sbuf = (__bf16*)smem;                       // [2][4][TILE_ELEMS]
  float*  red  = (float*)(smem + 2 * 4 * TILE_ELEMS * sizeof(__bf16));

  const int t    = threadIdx.x;
  const int lane = t & 31;
  const int wid  = t >> 5;       // 8 waves
  const int wr   = wid & 3;      // 4 row groups of 32
  const int wc   = wid >> 2;     // 2 col groups of 64
  const int bm   = blockIdx.y * BM;
  const int bn   = blockIdx.x * BN;

  const int fm = lane & 15;      // row within 16x16 tile
  const int kh = lane >> 4;      // K-half selector per WMMA layout

  // dst layout: [Ahi | Alo | Bhi | Blo], each TILE_ELEMS bf16
  auto load_panel = [&](int k0, __bf16* dst) {
    #pragma unroll
    for (int c = t; c < 512; c += 256) {
      const int r = c >> 2;
      const int q = c & 3;
      const size_t ga = (size_t)(bm + r) * DD + k0 + q * 8;
      const size_t gb = (size_t)(bn + r) * DD + k0 + q * 8;
      const int lofs = r * LDSW + q * 8;
      ot_cp16(&dst[lofs],                   &xhi[ga]);
      ot_cp16(&dst[TILE_ELEMS + lofs],      &xlo[ga]);
      ot_cp16(&dst[2 * TILE_ELEMS + lofs],  &yhi[gb]);
      ot_cp16(&dst[3 * TILE_ELEMS + lofs],  &ylo[gb]);
    }
  };

  v8f acc[2][4];
  #pragma unroll
  for (int i = 0; i < 2; ++i)
    #pragma unroll
    for (int j = 0; j < 4; ++j)
      #pragma unroll
      for (int e = 0; e < 8; ++e) acc[i][j][e] = 0.0f;

  // prologue: fill buffer 0
  load_panel(0, sbuf);
  ot_cp_wait();
  __syncthreads();

  for (int k0 = 0; k0 < DD; k0 += BK) {
    const int cur = (k0 >> 5) & 1;
    __bf16* cb = sbuf + cur * (4 * TILE_ELEMS);
    __bf16* nb = sbuf + (cur ^ 1) * (4 * TILE_ELEMS);

    // issue next panel's copies before touching the matrix pipe
    if (k0 + BK < DD) load_panel(k0 + BK, nb);

    __bf16* cAhi = cb;
    __bf16* cAlo = cb + TILE_ELEMS;
    __bf16* cBhi = cb + 2 * TILE_ELEMS;
    __bf16* cBlo = cb + 3 * TILE_ELEMS;

    // ---- LDS -> fragments (per documented 16-bit A/B VGPR layouts) ----
    v16bf ahi[2], alo[2], bhi[4], blo[4];
    #pragma unroll
    for (int mt = 0; mt < 2; ++mt) {
      const int base = (wr * 32 + mt * 16 + fm) * LDSW;
      const int c0 = base + kh * 8;        // K = kh*8 .. +7
      const int c1 = base + 16 + kh * 8;   // K = 16 + kh*8 .. +7
      ahi[mt] = combine8(*(const v8bf*)&cAhi[c0], *(const v8bf*)&cAhi[c1]);
      alo[mt] = combine8(*(const v8bf*)&cAlo[c0], *(const v8bf*)&cAlo[c1]);
    }
    #pragma unroll
    for (int nt = 0; nt < 4; ++nt) {
      const int base = (wc * 64 + nt * 16 + fm) * LDSW + kh * 8; // K = kh*8..+7, 16+kh*8..+7
      bhi[nt] = combine8(*(const v8bf*)&cBhi[base], *(const v8bf*)&cBhi[base + 16]);
      blo[nt] = combine8(*(const v8bf*)&cBlo[base], *(const v8bf*)&cBlo[base + 16]);
    }

    // ---- bf16x3: hi*hi + hi*lo + lo*hi, f32 accumulate ----
    #pragma unroll
    for (int mt = 0; mt < 2; ++mt) {
      #pragma unroll
      for (int nt = 0; nt < 4; ++nt) {
        acc[mt][nt] = __builtin_amdgcn_wmma_f32_16x16x32_bf16(
            false, ahi[mt], false, bhi[nt], (short)0, acc[mt][nt], false, false);
        acc[mt][nt] = __builtin_amdgcn_wmma_f32_16x16x32_bf16(
            false, ahi[mt], false, blo[nt], (short)0, acc[mt][nt], false, false);
        acc[mt][nt] = __builtin_amdgcn_wmma_f32_16x16x32_bf16(
            false, alo[mt], false, bhi[nt], (short)0, acc[mt][nt], false, false);
      }
    }

    ot_cp_wait();     // my async copies into nb have landed
    __syncthreads();  // everyone done reading cb / writing nb
  }

  // ---- fused epilogue: term = s - eps*exp((s - c)/eps) ----
  // C/D layout: element e of v8f -> M = e + 8*kh, N = lane & 15
  float local = 0.0f;
  #pragma unroll
  for (int mt = 0; mt < 2; ++mt) {
    #pragma unroll
    for (int nt = 0; nt < 4; ++nt) {
      const int jcol = bn + wc * 64 + nt * 16 + fm;
      const float vj  = vv[jcol];
      const float y2j = y2[jcol];
      #pragma unroll
      for (int e = 0; e < 8; ++e) {
        const int irow = bm + wr * 32 + mt * 16 + e + 8 * kh;
        const float g  = acc[mt][nt][e];
        const float s  = u[irow] + vj;
        const float c  = x2[irow] + y2j - 2.0f * g;
        local += s - OT_EPS * __expf((s - c) * OT_INV_EPS);
      }
    }
  }
  red[t] = local;
  __syncthreads();
  for (int o = 128; o > 0; o >>= 1) {
    if (t < o) red[t] += red[t + o];
    __syncthreads();
  }
  if (t == 0) partials[blockIdx.y * gridDim.x + blockIdx.x] = red[0];
}

__global__ __launch_bounds__(256)
void ot_finalize_kernel(const float* __restrict__ partials, int n, float* __restrict__ out)
{
  __shared__ float red[256];
  const int t = threadIdx.x;
  float s = 0.0f;
  for (int i = t; i < n; i += 256) s += partials[i];
  red[t] = s;
  __syncthreads();
  for (int o = 128; o > 0; o >>= 1) {
    if (t < o) red[t] += red[t + o];
    __syncthreads();
  }
  if (t == 0) out[0] = -(red[0] / (4096.0f * 4096.0f));
}

extern "C" void kernel_launch(void* const* d_in, const int* in_sizes, int n_in,
                              void* d_out, int out_size, void* d_ws, size_t ws_size,
                              hipStream_t stream) {
  const float* x   = (const float*)d_in[0];
  const float* y   = (const float*)d_in[1];
  const float* w_u = (const float*)d_in[2];
  const float* b_u = (const float*)d_in[3];
  const float* w_v = (const float*)d_in[4];
  const float* b_v = (const float*)d_in[5];

  char* ws = (char*)d_ws;
  const size_t S = (size_t)NROW * DD * sizeof(__bf16);  // 25,165,824 B per array
  __bf16* xhi = (__bf16*)(ws + 0 * S);
  __bf16* xlo = (__bf16*)(ws + 1 * S);
  __bf16* yhi = (__bf16*)(ws + 2 * S);
  __bf16* ylo = (__bf16*)(ws + 3 * S);
  float* u        = (float*)(ws + 4 * S);
  float* v        = (float*)(ws + 4 * S + 16384);
  float* x2       = (float*)(ws + 4 * S + 32768);
  float* y2       = (float*)(ws + 4 * S + 49152);
  float* partials = (float*)(ws + 4 * S + 65536);

  ot_prep_kernel<<<NROW, 256, 0, stream>>>(x, w_u, b_u, xhi, xlo, u, x2);
  ot_prep_kernel<<<NROW, 256, 0, stream>>>(y, w_v, b_v, yhi, ylo, v, y2);

  dim3 grid(NROW / BN, NROW / BM);  // 32 x 32
  ot_gemm_kernel<<<grid, 256, SHMEM_BYTES, stream>>>(xhi, xlo, yhi, ylo,
                                                     u, v, x2, y2, partials);

  ot_finalize_kernel<<<1, 256, 0, stream>>>(partials, (NROW / BN) * (NROW / BM),
                                            (float*)d_out);
}
